// BidirectionalCrossAttention_74071005987085
// MI455X (gfx1250) — matrix operand order repaired and checked by reading, not compiled, not run on hardware
//
#include <hip/hip_runtime.h>
#include <math.h>
#include <stdint.h>

typedef __bf16 bf16_t;
typedef __attribute__((ext_vector_type(16))) __bf16 v16bf;
typedef __attribute__((ext_vector_type(8)))  __bf16 v8bf;
typedef __attribute__((ext_vector_type(8)))  float  v8f;
typedef int v4i_vs __attribute__((vector_size(16)));

#define BATCH 8
#define CH    256
#define NSP   2304   // 48*48

struct PtrPack6 { const float* p[6]; };

// ---- CDNA5 async global->LDS copy (ASYNCcnt-tracked), with sync fallback ----
#if defined(__HIP_DEVICE_COMPILE__) && \
    __has_builtin(__builtin_amdgcn_global_load_async_to_lds_b128) && \
    __has_builtin(__builtin_amdgcn_s_wait_asynccnt)
#define USE_ASYNC_COPY 1
#else
#define USE_ASYNC_COPY 0
#endif

__device__ inline void copy16B_g2l(const bf16_t* g, bf16_t* l) {
#if USE_ASYNC_COPY
    // Builtin expects v4i* in AS1 (global) and AS3 (LDS).
    // flat->AS1 is value-identical; flat shared addr low 32 bits == LDS offset.
    typedef __attribute__((address_space(1))) v4i_vs* gp_t;
    typedef __attribute__((address_space(3))) v4i_vs* lp_t;
    __builtin_amdgcn_global_load_async_to_lds_b128(
        (gp_t)(unsigned long long)(uintptr_t)g,
        (lp_t)(unsigned int)(uintptr_t)l,
        0, 0);
#else
    *(v8bf*)l = *(const v8bf*)g;
#endif
}

__device__ inline void wait_async_copies() {
#if USE_ASYNC_COPY
    __builtin_amdgcn_s_wait_asynccnt(0);
#endif
}

// ---- WMMA helpers (CDNA5 gfx1250, wave32) ----

__device__ inline v8f wmma_bf16(v16bf a, v16bf b, v8f c) {
    // (neg_a, A, neg_b, B, c_mod, C, reuse_a, reuse_b)
    return __builtin_amdgcn_wmma_f32_16x16x32_bf16(false, a, false, b, (short)0, c, false, false);
}

// A-matrix 16x32 bf16 fragment: lane holds row m=lane&15.
// elements e=0..15 -> k = k0 + 16*half + e (contiguous 32B)
// `rowptr` points at element (m, k0 + 16*half).
__device__ inline v16bf load_frag_a(const bf16_t* rowptr) {
    return *(const v16bf*)(rowptr);
}

// B-matrix 32x16 bf16 fragment: lane holds col n=lane&15, k = 16*half + e,
// i.e. 16 contiguous bf16 (32 bytes) along the k axis.
__device__ inline v16bf load_frag_b(const bf16_t* colptr) {
    return *(const v16bf*)(colptr);
}

// ---- Kernel 1: x (f32 [b][c][n]) -> bf16 transposed [src][b][n][c] ----
__global__ void convert_x_kernel(const float* __restrict__ x1,
                                 const float* __restrict__ x2,
                                 bf16_t* __restrict__ xbT) {
    size_t i = (size_t)blockIdx.x * blockDim.x + threadIdx.x;
    const size_t total = 2ull * BATCH * CH * NSP;
    if (i >= total) return;
    int n = (int)(i % NSP); size_t r = i / NSP;
    int c = (int)(r % CH);  r /= CH;
    int b = (int)(r % BATCH);
    int src = (int)(r / BATCH);
    const float* x = src ? x2 : x1;
    float v = x[((size_t)b * CH + c) * NSP + n];
    xbT[(((size_t)src * BATCH + b) * NSP + n) * CH + c] = (bf16_t)v;
}

// ---- Kernel 2: weights f32 [co][ci] -> bf16 row-major, 6 packs ----
__global__ void convert_w_kernel(PtrPack6 w, bf16_t* __restrict__ Wbf) {
    int i = blockIdx.x * blockDim.x + threadIdx.x;
    const int total = 6 * CH * CH;
    if (i >= total) return;
    int p = i / (CH * CH);
    int e = i - p * (CH * CH);
    Wbf[i] = (bf16_t)w.p[p][e];
}

// ---- Kernel 3: projection GEMM  out[co][n] = W[co][ci] * xT[n][ci] + bias ----
// p = 0..5 (q1,k1,v1,q2,k2,v2). q/k stored transposed [b][n][co] (bf16),
// v stored natural [b][co][n] (bf16).
__global__ __launch_bounds__(256)
void proj_gemm_kernel(const bf16_t* __restrict__ Wbf,
                      const bf16_t* __restrict__ xbT,
                      PtrPack6 bias,
                      bf16_t* __restrict__ proj) {
    const int lane = threadIdx.x & 31;
    const int wave = threadIdx.x >> 5;
    const int half = lane >> 4;
    const int nl   = lane & 15;

    int gid = blockIdx.x * 8 + wave;          // one 16x16 tile per wave
    const int NT = NSP / 16;                  // 144
    int nt = gid % NT; int r = gid / NT;
    int cot = r % (CH / 16); r /= (CH / 16);
    int b = r % BATCH; int p = r / BATCH;     // 0..5
    int n0 = nt * 16, co0 = cot * 16;
    int src = (p < 3) ? 0 : 1;

    const bf16_t* Arow = Wbf + ((size_t)(p * CH + co0 + nl)) * CH + half * 16;
    const bf16_t* Bcol = xbT + (((size_t)(src * BATCH + b)) * NSP + n0 + nl) * CH + half * 16;

    v8f acc = {};
#pragma unroll
    for (int kc = 0; kc < CH; kc += 32) {
        v16bf a = load_frag_a(Arow + kc);
        v16bf bb = load_frag_b(Bcol + kc);
        acc = wmma_bf16(a, bb, acc);
    }
    const float* bp = bias.p[p];
#pragma unroll
    for (int v = 0; v < 8; ++v) acc[v] += bp[co0 + v + 8 * half];

    const size_t PS = (size_t)BATCH * NSP * CH;
    bf16_t* dst = proj + (size_t)p * PS;
    if (p == 2 || p == 5) {
        // V: natural [b][co][n]
#pragma unroll
        for (int v = 0; v < 8; ++v) {
            int co = co0 + v + 8 * half;
            dst[((size_t)b * CH + co) * NSP + n0 + nl] = (bf16_t)acc[v];
        }
    } else {
        // Q/K: transposed [b][n][co], 8 consecutive co -> one 16B store
        v8bf pk;
#pragma unroll
        for (int v = 0; v < 8; ++v) pk[v] = (bf16_t)acc[v];
        *(v8bf*)(dst + ((size_t)b * NSP + n0 + nl) * CH + co0 + 8 * half) = pk;
    }
}

// ---- Kernel 4: fused flash attention + residual, LDS-staged K/V ----
// dir 0: out1 = x1 + softmax(Q1^T K2) V2 ;  dir 1: out2 = x2 + softmax(Q2^T K1) V1
__global__ __launch_bounds__(256, 1)
void flash_attn_kernel(const bf16_t* __restrict__ proj,
                       const float* __restrict__ x1,
                       const float* __restrict__ x2,
                       float* __restrict__ out) {
    const size_t PS = (size_t)BATCH * NSP * CH;
    const int dir = blockIdx.z;
    const int b   = blockIdx.y;
    const int it  = blockIdx.x;            // i-tile of 128 rows
    const int tid  = threadIdx.x;
    const int wave = tid >> 5;
    const int lane = tid & 31;
    const int half = lane >> 4;
    const int nl   = lane & 15;

    const bf16_t* Qt = proj + (size_t)(dir ? 3 : 0) * PS + (size_t)b * NSP * CH; // [n][c]
    const bf16_t* Kt = proj + (size_t)(dir ? 1 : 4) * PS + (size_t)b * NSP * CH; // [n][c]
    const bf16_t* V  = proj + (size_t)(dir ? 2 : 5) * PS + (size_t)b * CH * NSP; // [c][n]
    const float*  x  = (dir ? x2 : x1) + (size_t)b * CH * NSP;
    float*        o  = out + (size_t)dir * BATCH * CH * NSP + (size_t)b * CH * NSP;

    const int i0 = it * 128 + wave * 16;   // this wave's 16 query rows

    // Double-buffered K/V tiles + per-wave P tile.  144 KB total (<320 KB WGP LDS).
    __shared__ __align__(32) bf16_t Ks[2][64][CH];   // [j_local][c]
    __shared__ __align__(32) bf16_t Vs[2][CH][64];   // [c][j_local]
    __shared__ __align__(32) bf16_t Pbuf[8][16][64];
    bf16_t (*P)[64] = Pbuf[wave];

    // Cooperative tile fill: K tile is 32 KB contiguous in [n][c] layout;
    // V tile is 256 rows of 128 B (stride NSP*2 in global).
    auto stage = [&](int buf, int j0) {
#pragma unroll
        for (int s = 0; s < 8; ++s) {
            int off = (s * 256 + tid) * 8;                 // bf16 elements
            copy16B_g2l(Kt + (size_t)j0 * CH + off, &Ks[buf][0][0] + off);
        }
#pragma unroll
        for (int s = 0; s < 8; ++s) {
            int idx = s * 256 + tid;                       // 0..2047 16B-chunks
            int c = idx >> 3, jb = (idx & 7) * 8;
            copy16B_g2l(V + (size_t)c * NSP + j0 + jb, &Vs[buf][c][jb]);
        }
    };

    v8f zero = {};
    v8f acc[16];                           // out^T tile: 16 rows(i) x 256 cols(c)
#pragma unroll
    for (int t = 0; t < 16; ++t) acc[t] = zero;
    float mrow[8], lrow[8];
#pragma unroll
    for (int v = 0; v < 8; ++v) { mrow[v] = -INFINITY; lrow[v] = 0.0f; }

    const bf16_t* Qrow = Qt + ((size_t)(i0 + nl)) * CH + half * 16;
    const int NT = NSP / 64;               // 36 j-tiles

    stage(0, 0);
    for (int jt = 0; jt < NT; ++jt) {
        const int buf = jt & 1;
        wait_async_copies();               // own async fills of `buf` complete
        __syncthreads();                   // all waves' fills done; buf^1 free
        if (jt + 1 < NT) stage(buf ^ 1, (jt + 1) * 64);  // prefetch next tile

        // ---- S = Q^T K over a 16(i) x 64(j) tile, K from LDS ----
        v8f S[4];
#pragma unroll
        for (int s = 0; s < 4; ++s) S[s] = zero;
#pragma unroll
        for (int kc = 0; kc < CH; kc += 32) {
            v16bf qa = load_frag_a(Qrow + kc);
#pragma unroll
            for (int s = 0; s < 4; ++s)
                S[s] = wmma_bf16(qa, load_frag_b(&Ks[buf][s * 16 + nl][kc + 16 * half]), S[s]);
        }

        // ---- online softmax (row m = v + 8*half lives across 16 lanes of a half) ----
        float t8[8];
#pragma unroll
        for (int v = 0; v < 8; ++v)
            t8[v] = fmaxf(fmaxf(S[0][v], S[1][v]), fmaxf(S[2][v], S[3][v]));
#pragma unroll
        for (int off = 1; off < 16; off <<= 1)
#pragma unroll
            for (int v = 0; v < 8; ++v)
                t8[v] = fmaxf(t8[v], __shfl_xor(t8[v], off, 32));

        float scale[8];
#pragma unroll
        for (int v = 0; v < 8; ++v) {
            float mn = fmaxf(mrow[v], t8[v]);
            scale[v] = __expf(mrow[v] - mn);
            mrow[v] = mn;
        }

        float rsum[8];
#pragma unroll
        for (int v = 0; v < 8; ++v) rsum[v] = 0.0f;
#pragma unroll
        for (int s = 0; s < 4; ++s)
#pragma unroll
            for (int v = 0; v < 8; ++v) {
                float pv = __expf(S[s][v] - mrow[v]);
                rsum[v] += pv;
                P[v + 8 * half][s * 16 + nl] = (bf16_t)pv;   // row-major P, A-frag friendly
            }
#pragma unroll
        for (int off = 1; off < 16; off <<= 1)
#pragma unroll
            for (int v = 0; v < 8; ++v)
                rsum[v] += __shfl_xor(rsum[v], off, 32);
#pragma unroll
        for (int v = 0; v < 8; ++v) lrow[v] = lrow[v] * scale[v] + rsum[v];

        // rescale running accumulator
#pragma unroll
        for (int t = 0; t < 16; ++t)
#pragma unroll
            for (int v = 0; v < 8; ++v) acc[t][v] *= scale[v];

        // ---- acc += P(16i x 32j) * V^T(32j x 16c), V from LDS ----
#pragma unroll
        for (int jc = 0; jc < 64; jc += 32) {
            v16bf pa = load_frag_a(&P[nl][jc + 16 * half]);
#pragma unroll
            for (int ct = 0; ct < 16; ++ct)
                acc[ct] = wmma_bf16(pa, load_frag_b(&Vs[buf][ct * 16 + nl][jc + 16 * half]), acc[ct]);
        }
    }

    // ---- epilogue: divide by row sum, add residual, store out[c][i] ----
#pragma unroll
    for (int v = 0; v < 8; ++v) lrow[v] = 1.0f / lrow[v];
#pragma unroll
    for (int ct = 0; ct < 16; ++ct) {
        int c = ct * 16 + nl;
#pragma unroll
        for (int v = 0; v < 8; ++v) {
            int i = i0 + v + 8 * half;
            size_t idx = (size_t)c * NSP + i;
            o[idx] = x[idx] + acc[ct][v] * lrow[v];
        }
    }
}

extern "C" void kernel_launch(void* const* d_in, const int* in_sizes, int n_in,
                              void* d_out, int out_size, void* d_ws, size_t ws_size,
                              hipStream_t stream) {
    const float* x1 = (const float*)d_in[0];
    const float* x2 = (const float*)d_in[1];
    PtrPack6 wv, bv;
    for (int p = 0; p < 6; ++p) {
        wv.p[p] = (const float*)d_in[2 + 2 * p];
        bv.p[p] = (const float*)d_in[3 + 2 * p];
    }

    char* ws = (char*)d_ws;
    size_t off = 0;
    bf16_t* Wbf = (bf16_t*)(ws + off);
    off += (size_t)6 * CH * CH * sizeof(bf16_t);
    off = (off + 255) & ~(size_t)255;
    bf16_t* xbT = (bf16_t*)(ws + off);
    off += (size_t)2 * BATCH * NSP * CH * sizeof(bf16_t);
    off = (off + 255) & ~(size_t)255;
    bf16_t* proj = (bf16_t*)(ws + off);
    // proj uses 6 * BATCH * NSP * CH * 2 bytes

    {   // weights -> bf16
        int total = 6 * CH * CH;
        convert_w_kernel<<<(total + 255) / 256, 256, 0, stream>>>(wv, Wbf);
    }
    {   // x -> bf16 transposed
        size_t total = (size_t)2 * BATCH * CH * NSP;
        convert_x_kernel<<<(unsigned)((total + 255) / 256), 256, 0, stream>>>(x1, x2, xbT);
    }
    {   // 6 projections: 6*8*(256/16)*(2304/16) = 110592 wave-tiles / 8 waves per block
        int blocks = 6 * BATCH * (CH / 16) * (NSP / 16) / 8;
        proj_gemm_kernel<<<blocks, 256, 0, stream>>>(Wbf, xbT, bv, proj);
    }
    {   // fused flash attention, both directions
        dim3 grid(NSP / 128, BATCH, 2);
        flash_attn_kernel<<<grid, 256, 0, stream>>>(proj, x1, x2, (float*)d_out);
    }
}
